// CustomMultiheadSelfAttention_77137612636651
// MI455X (gfx1250) — matrix operand order repaired and checked by reading, not compiled, not run on hardware
//
#include <hip/hip_runtime.h>
#include <hip/hip_bf16.h>

// ---------------------------------------------------------------------------
// CustomMultiheadSelfAttention for MI455X (gfx1250), wave32 + WMMA bf16
// + Tensor Data Mover (TDM) staging of GEMM B-operands into LDS.
//   B=4, S=1024, E=1024, H=16, DK=64.
//   d_out = [attn_output (4M f32) | attn_weights (64M f32)].
// ---------------------------------------------------------------------------

typedef __bf16 bf16;
typedef __attribute__((ext_vector_type(16))) __bf16 v16bf;
typedef __attribute__((ext_vector_type(8)))  __bf16 v8bf;
typedef __attribute__((ext_vector_type(8)))  float  v8f;
typedef __attribute__((ext_vector_type(4)))  unsigned int v4u;
typedef __attribute__((ext_vector_type(8)))  int  v8i;
typedef __attribute__((ext_vector_type(4)))  int  v4i;

#define B_   4
#define S_   1024
#define E_   1024
#define H_   16
#define DK_  64
#define N3E_ 3072
#define NEGV (-10000.0f)

union U16 { v16bf v; v8bf h[2]; };

// A-operand (16x32 bf16, M x K): lane r=l&15 holds row r; half=l>>4.
// elements 0..15 = K(kbase+half*16+0..15)
static __device__ inline v16bf load_a16(const bf16* base) {
  U16 u;
  u.h[0] = *(const v8bf*)(base);
  u.h[1] = *(const v8bf*)(base + 8);
  return u.v;
}

// B-operand (32x16 bf16, K x N): lane n=l&15 holds column n; half=l>>4.
// elements 0..15 = K(kbase+half*16+0..15) -> contiguous along BT row n.
static __device__ inline v16bf load_b16(const bf16* p) {
  U16 u;
  u.h[0] = *(const v8bf*)(p);
  u.h[1] = *(const v8bf*)(p + 8);
  return u.v;
}

static __device__ inline v8f wmma_bf16(v16bf a, v16bf b, v8f c) {
  // 8 args: (neg_a, A, neg_b, B, c_mod, C, reuse_a, reuse_b)
  return __builtin_amdgcn_wmma_f32_16x16x32_bf16(false, a, false, b,
                                                 (short)0, c, false, false);
}

// ---------------------------------------------------------------------------
// TDM: stage a (64 rows x 32 cols) bf16 tile of a row-major (rows x 1024)
// matrix into LDS.  D# per CDNA5 ISA:
//   group0: [1:0]=count=1, [63:32]=lds_addr, [120:64]=global_addr, type=2
//   group1: data_size=1(2B), tensor_dim0=1024, tensor_dim1=rows,
//           tile_dim0=32, tile_dim1=64, tensor_dim0_stride=1024
// groups 2/3 zero (2D tensor).  Tracked with TENSORcnt.
// NOTE: this toolchain exposes the 6-arg clang-23 builtin:
//   (v4u g0, v8i g1, v4i g2, v4i g3, v8i g4, i32 cpol)
// ---------------------------------------------------------------------------
static __device__ inline void tdm_load_tile(unsigned lds_off,
                                            unsigned long long gaddr,
                                            unsigned rows_total) {
  v4u g0;
  g0[0] = 1u;                                           // count = 1
  g0[1] = lds_off;                                      // lds_addr (bytes)
  g0[2] = (unsigned)(gaddr & 0xffffffffu);              // global_addr[31:0]
  g0[3] = (unsigned)((gaddr >> 32) & 0x1ffffffu)        // global_addr[56:32]
          | (2u << 30);                                 // type = 2 (image)
  v8i g1;
  g1[0] = (int)(1u << 16);                              // data_size=1 (2B)
  g1[1] = (int)(1024u << 16);                           // tensor_dim0 = 1024
  g1[2] = (int)((rows_total & 0xffffu) << 16);          // tensor_dim1 = rows
  g1[3] = (int)(32u << 16);                             // tile_dim0 = 32
  g1[4] = (int)64;                                      // tile_dim1 = 64
  g1[5] = (int)1024;                                    // dim0_stride = 1024
  g1[6] = 0;
  g1[7] = 0;
  v4i z4 = {0, 0, 0, 0};
  v8i z8 = {0, 0, 0, 0, 0, 0, 0, 0};
  __builtin_amdgcn_tensor_load_to_lds(g0, g1, z4, z4, z8, 0);
}

// ---------------------------------------------------------------------------
// Shared 64x64-per-wave GEMM mainloop with TDM double-buffered B staging.
//   Abase  : per-lane A row pointer (row tileM*64 + r), stride E_
//   wtile  : byte address of W tile start (row tileN*64, col 0), row-major K=1024
//   sB     : LDS double buffer [2][64][32] bf16 (8 KB)
// All 8 waves of the block share the same B tile; wave 0 drives the TDM.
// ---------------------------------------------------------------------------
static __device__ inline void gemm64x64_tdm(const bf16* __restrict__ Abase,
                                            unsigned long long wtile,
                                            unsigned rows_total,
                                            bf16* sB, int lane,
                                            v8f (&c)[4][4]) {
  const int r    = lane & 15;
  const int half = lane >> 4;
  const bool issuer = (threadIdx.x < 32);
  const unsigned lds0 = (unsigned)(uintptr_t)sB;

  if (issuer) tdm_load_tile(lds0, wtile, rows_total);

  for (int kb = 0; kb < E_; kb += 32) {
    const int buf = (kb >> 5) & 1;
    if (issuer) {
      if (kb + 32 < E_) {
        tdm_load_tile(lds0 + (unsigned)((buf ^ 1) * 4096),
                      wtile + (unsigned long long)(kb + 32) * 2, rows_total);
        __builtin_amdgcn_s_wait_tensorcnt(1);   // oldest (current buf) done
      } else {
        __builtin_amdgcn_s_wait_tensorcnt(0);
      }
    }
    __syncthreads();   // B tile for this kb visible to all waves

    const bf16* Bl = sB + buf * (64 * 32);
    v16bf a[4], b[4];
#pragma unroll
    for (int i = 0; i < 4; ++i)
      a[i] = load_a16(Abase + (size_t)i * 16 * E_ + kb + half * 16);
#pragma unroll
    for (int i = 0; i < 4; ++i)
      b[i] = load_b16(Bl + (i * 16 + r) * 32 + half * 16);
#pragma unroll
    for (int mi = 0; mi < 4; ++mi)
#pragma unroll
      for (int ni = 0; ni < 4; ++ni)
        c[mi][ni] = wmma_bf16(a[mi], b[ni], c[mi][ni]);

    __syncthreads();   // everyone done with buf before TDM overwrites it
  }
}

// ---------------------------------------------------------------------------
// Kernel 0: f32 -> bf16 conversion (grid-stride)
// ---------------------------------------------------------------------------
__global__ void cvt_f32_bf16(const float* __restrict__ src,
                             bf16* __restrict__ dst, int n) {
  int i = blockIdx.x * blockDim.x + threadIdx.x;
  int stride = gridDim.x * blockDim.x;
  for (; i < n; i += stride) dst[i] = (bf16)src[i];
}

// ---------------------------------------------------------------------------
// Kernel 1: QKV projection.  qkv = x @ qkv_w^T + qkv_b
//   M = 4096 (64 tiles), N = 3072 (48 tiles), K = 1024.  64x64 per wave.
//   B tile staged via TDM (all 8 waves in a block share tileN).
//   Epilogue: bias, key-padding zeroing, split into q (x0.125), k, v^T (bf16).
// ---------------------------------------------------------------------------
__global__ __launch_bounds__(256)
void qkv_gemm(const bf16* __restrict__ xb, const bf16* __restrict__ wb,
              const float* __restrict__ bias, const int* __restrict__ kpm,
              bf16* __restrict__ q, bf16* __restrict__ k,
              bf16* __restrict__ vT) {
  __shared__ __align__(16) bf16 sB[2][64][32];   // 8 KB double buffer

  const int wave  = blockIdx.x * (blockDim.x >> 5) + (threadIdx.x >> 5);
  const int lane  = threadIdx.x & 31;
  const int tileM = wave & 63;   // 64 M-tiles (differs per wave)
  const int tileN = wave >> 6;   // 48 N-tiles (uniform within block)
  const int r     = lane & 15;
  const int half  = lane >> 4;

  v8f c[4][4] = {};

  const bf16* Abase = xb + (size_t)(tileM * 64 + r) * E_;
  const unsigned long long wtile =
      (unsigned long long)(uintptr_t)(wb + (size_t)(tileN * 64) * E_);

  gemm64x64_tdm(Abase, wtile, N3E_, &sB[0][0][0], lane, c);

#pragma unroll
  for (int mi = 0; mi < 4; ++mi) {
#pragma unroll
    for (int ni = 0; ni < 4; ++ni) {
      const int n_g   = tileN * 64 + ni * 16 + r;
      const int chunk = n_g >> 10;        // 0=q 1=k 2=v
      const int e     = n_g & 1023;
      const int h     = e >> 6;
      const int d     = e & 63;
      const float bv  = bias[n_g];
#pragma unroll
      for (int i = 0; i < 8; ++i) {
        const int m_g = tileM * 64 + mi * 16 + half * 8 + i;
        const int bi  = m_g >> 10;
        const int si  = m_g & 1023;
        float val = c[mi][ni][i] + bv;
        if (kpm[m_g] != 0) val = 0.0f;     // zero q/k/v at padded tokens
        if (chunk == 0) {
          q[(size_t)((bi * H_ + h) * S_ + si) * DK_ + d] = (bf16)(val * 0.125f);
        } else if (chunk == 1) {
          k[(size_t)((bi * H_ + h) * S_ + si) * DK_ + d] = (bf16)val;
        } else {
          vT[(size_t)((bi * H_ + h) * DK_ + d) * S_ + si] = (bf16)val;
        }
      }
    }
  }
}

// ---------------------------------------------------------------------------
// Kernel 2: attention.  One wave (block of 32) per (b, h, 16-query-row tile).
//   scores = q@k^T (q pre-scaled) -> LDS f32, +causal, +key-pad NEG
//   row softmax (f32), normalized weights -> d_out (f32) and LDS bf16 P
//   attn = P @ v  (B operand is v^T, contiguous)  -> bf16 workspace
// ---------------------------------------------------------------------------
__global__ __launch_bounds__(32)
void attn_kernel(const bf16* __restrict__ q, const bf16* __restrict__ kk,
                 const bf16* __restrict__ vT, const int* __restrict__ kpm,
                 float* __restrict__ weights, bf16* __restrict__ attnb) {
  __shared__ __align__(16) float sS[16 * S_];   // 64 KB score tile
  __shared__ __align__(16) bf16  sP[16 * S_];   // 32 KB bf16 P tile

  const int lane  = threadIdx.x & 31;
  const int r     = lane & 15;
  const int half  = lane >> 4;
  const int qt    = blockIdx.x & (S_ / 16 - 1); // 64 query tiles
  const int bh    = blockIdx.x >> 6;            // 0..63
  const int b     = bh >> 4;
  const int h     = bh & 15;
  const int qrow0 = qt * 16;

  const bf16* Qb = q  + (size_t)(bh * S_ + qrow0 + r) * DK_;
  const bf16* Kb = kk + (size_t)bh * S_ * DK_;

  // ---- scores: 64 key tiles x (DK=64 -> 2 wmma each) -----------------------
  for (int kt = 0; kt < S_ / 16; ++kt) {
    v8f c = {};
    const bf16* Krow = Kb + (size_t)(kt * 16 + r) * DK_;
#pragma unroll
    for (int kb2 = 0; kb2 < DK_; kb2 += 32) {
      v16bf a  = load_a16(Qb   + kb2 + half * 16);
      v16bf bb = load_b16(Krow + kb2 + half * 16);
      c = wmma_bf16(a, bb, c);
    }
    const int  col  = kt * 16 + r;
    const bool padk = kpm[b * S_ + col] != 0;
#pragma unroll
    for (int i = 0; i < 8; ++i) {
      const int row = half * 8 + i;
      float sc = c[i];
      if (col > qrow0 + row) sc += NEGV;   // causal (additive, like reference)
      if (padk)              sc  = NEGV;   // key padding (replace)
      sS[row * S_ + col] = sc;
    }
  }
  __syncthreads();

  // ---- softmax: lane handles row r, columns [half*512, half*512+512) -------
  {
    float* rowp = sS + r * S_ + half * 512;
    float mx = -3.0e38f;
    for (int j = 0; j < 512; ++j) mx = fmaxf(mx, rowp[j]);
    mx = fmaxf(mx, __shfl_xor(mx, 16, 32));
    float sum = 0.0f;
    for (int j = 0; j < 512; ++j) {
      float e = __expf(rowp[j] - mx);
      rowp[j] = e;
      sum += e;
    }
    sum += __shfl_xor(sum, 16, 32);
    const float inv = 1.0f / sum;
    float* gw = weights + (size_t)(bh * S_ + qrow0 + r) * S_ + half * 512;
    bf16*  pp = sP + r * S_ + half * 512;
    for (int j = 0; j < 512; ++j) {
      float w = rowp[j] * inv;
      gw[j] = w;             // full attn_weights to d_out (f32)
      pp[j] = (bf16)w;       // bf16 P for the P@V matmul
    }
  }
  __syncthreads();

  // ---- attn = P @ V : K = S = 1024, N = DK = 64 (4 tiles) ------------------
  const bf16* Vb = vT + (size_t)bh * DK_ * S_;
  v8f o[4] = {};
  for (int kb2 = 0; kb2 < S_; kb2 += 32) {
    v16bf a = load_a16(sP + r * S_ + kb2 + half * 16);
#pragma unroll
    for (int ni = 0; ni < 4; ++ni) {
      v16bf bb = load_b16(Vb + (size_t)(ni * 16 + r) * S_ + kb2 + half * 16);
      o[ni] = wmma_bf16(a, bb, o[ni]);
    }
  }
#pragma unroll
  for (int ni = 0; ni < 4; ++ni) {
    const int colE = h * 64 + ni * 16 + r;
#pragma unroll
    for (int i = 0; i < 8; ++i) {
      const int s = qrow0 + half * 8 + i;
      attnb[(size_t)(b * S_ + s) * E_ + colE] = (bf16)o[ni][i];
    }
  }
}

// ---------------------------------------------------------------------------
// Kernel 3: output projection.  out = attn @ out_w^T + out_b  (f32 out)
//   M = 4096 (64 tiles), N = 1024 (16 tiles), K = 1024.  TDM-staged B tile.
// ---------------------------------------------------------------------------
__global__ __launch_bounds__(256)
void out_proj(const bf16* __restrict__ attnb, const bf16* __restrict__ wob,
              const float* __restrict__ ob, float* __restrict__ out) {
  __shared__ __align__(16) bf16 sB[2][64][32];

  const int wave  = blockIdx.x * (blockDim.x >> 5) + (threadIdx.x >> 5);
  const int lane  = threadIdx.x & 31;
  const int tileM = wave & 63;   // 64 M-tiles
  const int tileN = wave >> 6;   // 16 N-tiles (uniform within block)
  const int r     = lane & 15;
  const int half  = lane >> 4;

  v8f c[4][4] = {};

  const bf16* Abase = attnb + (size_t)(tileM * 64 + r) * E_;
  const unsigned long long wtile =
      (unsigned long long)(uintptr_t)(wob + (size_t)(tileN * 64) * E_);

  gemm64x64_tdm(Abase, wtile, E_, &sB[0][0][0], lane, c);

#pragma unroll
  for (int mi = 0; mi < 4; ++mi) {
#pragma unroll
    for (int ni = 0; ni < 4; ++ni) {
      const int n_g  = tileN * 64 + ni * 16 + r;
      const float bv = ob[n_g];
#pragma unroll
      for (int i = 0; i < 8; ++i) {
        const int m_g = tileM * 64 + mi * 16 + half * 8 + i;
        out[(size_t)m_g * E_ + n_g] = c[mi][ni][i] + bv;
      }
    }
  }
}

// ---------------------------------------------------------------------------
// Launch
// ---------------------------------------------------------------------------
extern "C" void kernel_launch(void* const* d_in, const int* in_sizes, int n_in,
                              void* d_out, int out_size, void* d_ws,
                              size_t ws_size, hipStream_t stream) {
  const float* x      = (const float*)d_in[0];
  // d_in[1] = attn_mask (causal) -- recomputed analytically in-kernel
  const int*   kpm    = (const int*)d_in[2];   // key_padding_mask (B,S)
  const float* qkv_w  = (const float*)d_in[3];
  const float* qkv_b  = (const float*)d_in[4];
  const float* out_w  = (const float*)d_in[5];
  const float* out_b  = (const float*)d_in[6];

  float* out     = (float*)d_out;                        // (B,S,E) f32
  float* weights = out + (size_t)B_ * S_ * E_;           // (B,H,S,S) f32

  char* ws = (char*)d_ws;
  bf16* xb    = (bf16*)(ws);                             //  8 MB  x bf16
  bf16* wqkvb = (bf16*)(ws + ((size_t)8  << 20));        //  6 MB  qkv_w bf16
  bf16* wob   = (bf16*)(ws + ((size_t)14 << 20));        //  2 MB  out_w bf16
  bf16* qv    = (bf16*)(ws + ((size_t)16 << 20));        //  8 MB  q (B,H,S,DK)
  bf16* kv    = (bf16*)(ws + ((size_t)24 << 20));        //  8 MB  k (B,H,S,DK)
  bf16* vT    = (bf16*)(ws + ((size_t)32 << 20));        //  8 MB  v^T (B,H,DK,S)
  bf16* attnb = (bf16*)(ws + ((size_t)40 << 20));        //  8 MB  attn (B,S,E)

  cvt_f32_bf16<<<1024, 256, 0, stream>>>(x,     xb,    B_ * S_ * E_);
  cvt_f32_bf16<<<768,  256, 0, stream>>>(qkv_w, wqkvb, N3E_ * E_);
  cvt_f32_bf16<<<256,  256, 0, stream>>>(out_w, wob,   E_ * E_);

  // 64 M-tiles * 48 N-tiles = 3072 waves, 8 waves/block (tileN uniform/block)
  qkv_gemm<<<384, 256, 0, stream>>>(xb, wqkvb, qkv_b, kpm, qv, kv, vT);

  // (B*H) * (S/16) = 64 * 64 = 4096 single-wave blocks
  attn_kernel<<<4096, 32, 0, stream>>>(qv, kv, vT, kpm, weights, attnb);

  // 64 M-tiles * 16 N-tiles = 1024 waves, 8 waves/block
  out_proj<<<128, 256, 0, stream>>>(attnb, wob, out_b, out);
}
